// HybridSparseGATModel_24670292148960
// MI455X (gfx1250) — matrix operand order repaired and checked by reading, not compiled, not run on hardware
//
#include <hip/hip_runtime.h>
#include <cmath>

typedef __attribute__((ext_vector_type(16))) _Float16 v16h;
typedef __attribute__((ext_vector_type(8)))  float    v8f;

#define N_AGENTS 7
#define HID      16
#define NF       32
#define CD       16
#define GD       128   // 7*16 + 16
#define NEG_INF  (-1.0e9f)

struct Params {
  const float *node, *ctx, *base, *regime;
  const float *edge, *W1, *a1, *W2, *a2;
  const float *bh_w1, *bh_b1, *bh_w2, *bh_b2;
  const float *rh_w1, *rh_b1, *rh_w2, *rh_b2;
  const float *g_w1,  *g_b1,  *g_w2,  *g_b2;
  const float *res_w1,*res_b1,*res_w2,*res_b2;
  float* out;
  int B;
};

// K index of half `i` (0..15) in the 16-bit A-matrix fragment (16x32) for
// lane-half `hi` (0 = lanes 0-15, 1 = lanes 16-31), contiguous order:
// lanes0-15 -> K0-15, lanes16-31 -> K16-31 (matches the B fragments).
__device__ __forceinline__ int kof(int i, int hi) {
  return (hi << 4) + i;
}

__device__ __forceinline__ float sigmoidf_(float x) {
  return 1.0f / (1.0f + __expf(-x));
}

// Per-element GAT attention. Two lanes cooperate per element:
// half 0 handles rows 0..3, half 1 handles rows 4..6.
// hin: (7x16) raw projected features for this wave's 16 elements (LDS).
// hout: relu(alpha @ hin) written back (LDS).
__device__ void gat_attention(const float* __restrict__ hin,
                              float* __restrict__ hout,
                              const float* __restrict__ smask,
                              const float* __restrict__ a_vec,
                              int lane) {
  const int e    = lane >> 1;
  const int half = lane & 1;
  const int r0   = half ? 4 : 0;
  const int r1   = half ? 7 : 4;

  float ai[HID], aj[HID];
#pragma unroll
  for (int d = 0; d < HID; ++d) { ai[d] = a_vec[d]; aj[d] = a_vec[HID + d]; }

  const float* hb = hin + e * (N_AGENTS * HID);

  float sj[N_AGENTS];
#pragma unroll
  for (int j = 0; j < N_AGENTS; ++j) {
    float s = 0.0f;
#pragma unroll
    for (int d = 0; d < HID; ++d) s += hb[j * HID + d] * aj[d];
    sj[j] = s;
  }

  for (int i = r0; i < r1; ++i) {
    float si = 0.0f;
#pragma unroll
    for (int d = 0; d < HID; ++d) si += hb[i * HID + d] * ai[d];

    float ev[N_AGENTS], mx = -3.4e38f;
#pragma unroll
    for (int j = 0; j < N_AGENTS; ++j) {
      float v = si + sj[j];
      v = (v > 0.0f) ? v : 0.2f * v;          // leaky_relu(., 0.2)
      const float mk = smask[i * N_AGENTS + j];
      v = v * mk + (1.0f - mk) * NEG_INF;     // soft mask
      ev[j] = v;
      mx = fmaxf(mx, v);
    }
    float den = 0.0f;
#pragma unroll
    for (int j = 0; j < N_AGENTS; ++j) { ev[j] = __expf(ev[j] - mx); den += ev[j]; }
    const float inv = 1.0f / den;

    float acc[HID];
#pragma unroll
    for (int d = 0; d < HID; ++d) acc[d] = 0.0f;
#pragma unroll
    for (int j = 0; j < N_AGENTS; ++j) {
      const float al = ev[j] * inv;
#pragma unroll
      for (int d = 0; d < HID; ++d) acc[d] += al * hb[j * HID + d];
    }
#pragma unroll
    for (int d = 0; d < HID; ++d)
      hout[(i * HID + d) + e * (N_AGENTS * HID)] = fmaxf(acc[d], 0.0f);  // relu
  }
}

__global__ __launch_bounds__(128, 1)
void hybrid_gat_kernel(Params p) {
  __shared__ float sA[4][112 * HID];      // ping buffer per wave (7KB each)
  __shared__ float sB[4][112 * HID];      // pong buffer per wave
  __shared__ float sM[N_AGENTS * N_AGENTS];

  const int tid  = threadIdx.x;
  const int wave = tid >> 5;
  const int lane = tid & 31;
  const int hi   = (lane >> 4) & 1;       // lane half (0: lanes 0-15)
  const int n16  = lane & 15;
  const int eb   = (blockIdx.x * 4 + wave) * 16;  // first element of this wave's tile
  const int B    = p.B;

  float* bufA = sA[wave];
  float* bufB = sB[wave];

  // Edge mask: sigmoid(edge_logits), shared by all elements; also an output.
  if (tid < N_AGENTS * N_AGENTS) {
    const float m = sigmoidf_(p.edge[tid]);
    sM[tid] = m;
    if (blockIdx.x == 0) p.out[4 * (size_t)B + tid] = m;
  }

  // Warm caches for the per-element side inputs while GEMM1 streams node_feats.
  __builtin_prefetch(&p.ctx[(size_t)eb * CD], 0, 0);
  __builtin_prefetch(&p.regime[(size_t)eb * 3], 0, 0);
  __builtin_prefetch(&p.base[(size_t)eb * 2], 0, 0);

  // ---------------- GEMM1: h1 = node_feats @ W1  (112x32 @ 32x16) -----------
  {
    v16h bw1;                              // B fragment: W1 (32x16), broadcast
#pragma unroll
    for (int i = 0; i < 16; ++i)
      bw1[i] = (_Float16)p.W1[(hi * 16 + i) * HID + n16];

    v8f cfr[7];
#pragma unroll
    for (int t = 0; t < 7; ++t) {
      const long row = (long)eb * N_AGENTS + t * 16 + n16;  // contiguous rows
      v16h a;
#pragma unroll
      for (int i = 0; i < 16; ++i)
        a[i] = (_Float16)p.node[row * NF + kof(i, hi)];
      v8f c = {0.f, 0.f, 0.f, 0.f, 0.f, 0.f, 0.f, 0.f};
      cfr[t] = __builtin_amdgcn_wmma_f32_16x16x32_f16(
          false, a, false, bw1, (short)0, c, false, false);
    }
#pragma unroll
    for (int t = 0; t < 7; ++t)
#pragma unroll
      for (int cc = 0; cc < 8; ++cc)
        bufA[(t * 16 + cc + 8 * hi) * HID + n16] = cfr[t][cc];
  }
  __syncthreads();

  // ---------------- GAT layer 1 attention (sA -> relu -> sB) ----------------
  gat_attention(bufA, bufB, sM, p.a1, lane);
  __syncthreads();

  // ---------------- GEMM2: h2 = h @ W2 (112x16 @ 16x16, K padded to 32) -----
  {
    v16h bw2;
#pragma unroll
    for (int i = 0; i < 16; ++i)
      bw2[i] = (_Float16)((hi == 0) ? p.W2[i * HID + n16] : 0.0f);  // K>=16 -> 0

    v8f cfr[7];
#pragma unroll
    for (int t = 0; t < 7; ++t) {
      const int row = t * 16 + n16;
      v16h a;
#pragma unroll
      for (int i = 0; i < 16; ++i) {
        const int k = kof(i, hi);
        a[i] = (_Float16)((k < HID) ? bufB[row * HID + k] : 0.0f);
      }
      v8f c = {0.f, 0.f, 0.f, 0.f, 0.f, 0.f, 0.f, 0.f};
      cfr[t] = __builtin_amdgcn_wmma_f32_16x16x32_f16(
          false, a, false, bw2, (short)0, c, false, false);
    }
#pragma unroll
    for (int t = 0; t < 7; ++t)
#pragma unroll
      for (int cc = 0; cc < 8; ++cc)
        bufA[(t * 16 + cc + 8 * hi) * HID + n16] = cfr[t][cc];
  }
  __syncthreads();

  // ---------------- GAT layer 2 attention (sA -> relu -> sB) ----------------
  gat_attention(bufA, bufB, sM, p.a2, lane);
  __syncthreads();

  // ------- Head GEMM: combined(16x128) @ [bh_w1|rh_w1|res_w1](128x80) -------
  // combined[k] = graph_embed (k<112, in bufB as e*112+k) else context[k-112].
  {
    v16h afr[4];
#pragma unroll
    for (int kc = 0; kc < 4; ++kc) {
#pragma unroll
      for (int i = 0; i < 16; ++i) {
        const int k = kc * 32 + kof(i, hi);
        const float v = (k < 112)
            ? bufB[n16 * 112 + k]
            : p.ctx[(long)(eb + n16) * CD + (k - 112)];
        afr[kc][i] = (_Float16)v;
      }
    }
#pragma unroll
    for (int nt = 0; nt < 5; ++nt) {            // 80 output cols = 5 tiles
      v8f c = {0.f, 0.f, 0.f, 0.f, 0.f, 0.f, 0.f, 0.f};
#pragma unroll
      for (int kc = 0; kc < 4; ++kc) {          // 128 K = 4 chunks
        v16h b;
#pragma unroll
        for (int i = 0; i < 16; ++i) {
          const int K   = kc * 32 + hi * 16 + i;
          const int col = nt * 16 + n16;
          const float w = (nt < 2) ? p.bh_w1[K * 32 + col]
                        : (nt < 4) ? p.rh_w1[K * 32 + (col - 32)]
                                   : p.res_w1[K * 16 + (col - 64)];
          b[i] = (_Float16)w;
        }
        c = __builtin_amdgcn_wmma_f32_16x16x32_f16(
            false, afr[kc], false, b, (short)0, c, false, false);
      }
#pragma unroll
      for (int cc = 0; cc < 8; ++cc)
        bufA[(cc + 8 * hi) * 80 + nt * 16 + n16] = c[cc];  // stage 16x80
    }
  }
  __syncthreads();

  // ---------------- Epilogue: one element per lane (lanes 0-15) -------------
  if (lane < 16) {
    const int eg = eb + lane;
    const float* head = bufA + lane * 80;

    // Base head: softmax(relu(head[0:32]+b) @ bh_w2 + b2)
    float l0 = p.bh_b2[0], l1 = p.bh_b2[1];
#pragma unroll
    for (int j = 0; j < 32; ++j) {
      const float h = fmaxf(head[j] + p.bh_b1[j], 0.0f);
      l0 += h * p.bh_w2[j * 2 + 0];
      l1 += h * p.bh_w2[j * 2 + 1];
    }
    float m = fmaxf(l0, l1);
    float e0 = __expf(l0 - m), e1 = __expf(l1 - m);
    float inv = 1.0f / (e0 + e1);
    const float wb0 = e0 * inv, wb1 = e1 * inv;

    // Regime head
    l0 = p.rh_b2[0]; l1 = p.rh_b2[1];
#pragma unroll
    for (int j = 0; j < 32; ++j) {
      const float h = fmaxf(head[32 + j] + p.rh_b1[j], 0.0f);
      l0 += h * p.rh_w2[j * 2 + 0];
      l1 += h * p.rh_w2[j * 2 + 1];
    }
    m = fmaxf(l0, l1);
    e0 = __expf(l0 - m); e1 = __expf(l1 - m);
    inv = 1.0f / (e0 + e1);
    const float wr0 = e0 * inv, wr1 = e1 * inv;

    // Gate: sigmoid(tanh(regime @ g_w1 + b1) @ g_w2 + b2)
    const float r0 = p.regime[(size_t)eg * 3 + 0];
    const float r1 = p.regime[(size_t)eg * 3 + 1];
    const float r2 = p.regime[(size_t)eg * 3 + 2];
    float gacc = p.g_b2[0];
#pragma unroll
    for (int j = 0; j < 8; ++j) {
      const float t = r0 * p.g_w1[0 * 8 + j] + r1 * p.g_w1[1 * 8 + j] +
                      r2 * p.g_w1[2 * 8 + j] + p.g_b1[j];
      gacc += tanhf(t) * p.g_w2[j];
    }
    const float gate = sigmoidf_(gacc);

    // Residual: 0.05 * (tanh(head[64:80]+b) @ res_w2 + b2)
    float racc = p.res_b2[0];
#pragma unroll
    for (int j = 0; j < 16; ++j)
      racc += tanhf(head[64 + j] + p.res_b1[j]) * p.res_w2[j];
    const float residual = racc * 0.05f;

    const float w0 = (1.0f - gate) * wb0 + gate * wr0;
    const float w1 = (1.0f - gate) * wb1 + gate * wr1;
    float pred = w0 * p.base[(size_t)eg * 2 + 0] +
                 w1 * p.base[(size_t)eg * 2 + 1] + residual;
    pred = fmaxf(pred, 0.05f);

    p.out[eg]                        = pred;
    p.out[(size_t)B + eg * 2 + 0]    = w0;
    p.out[(size_t)B + eg * 2 + 1]    = w1;
    p.out[3 * (size_t)B + eg]        = gate;
  }
}

extern "C" void kernel_launch(void* const* d_in, const int* in_sizes, int n_in,
                              void* d_out, int out_size, void* d_ws, size_t ws_size,
                              hipStream_t stream) {
  (void)n_in; (void)out_size; (void)d_ws; (void)ws_size;
  Params p;
  p.node   = (const float*)d_in[0];
  p.ctx    = (const float*)d_in[1];
  p.base   = (const float*)d_in[2];
  p.regime = (const float*)d_in[3];
  p.edge   = (const float*)d_in[4];
  p.W1     = (const float*)d_in[5];
  p.a1     = (const float*)d_in[6];
  p.W2     = (const float*)d_in[7];
  p.a2     = (const float*)d_in[8];
  p.bh_w1  = (const float*)d_in[9];
  p.bh_b1  = (const float*)d_in[10];
  p.bh_w2  = (const float*)d_in[11];
  p.bh_b2  = (const float*)d_in[12];
  p.rh_w1  = (const float*)d_in[13];
  p.rh_b1  = (const float*)d_in[14];
  p.rh_w2  = (const float*)d_in[15];
  p.rh_b2  = (const float*)d_in[16];
  p.g_w1   = (const float*)d_in[17];
  p.g_b1   = (const float*)d_in[18];
  p.g_w2   = (const float*)d_in[19];
  p.g_b2   = (const float*)d_in[20];
  p.res_w1 = (const float*)d_in[21];
  p.res_b1 = (const float*)d_in[22];
  p.res_w2 = (const float*)d_in[23];
  p.res_b2 = (const float*)d_in[24];
  p.out    = (float*)d_out;
  p.B      = in_sizes[1] / CD;   // context is (B, 16)

  const int blocks = (p.B + 63) / 64;   // 64 elements per 128-thread block
  hipLaunchKernelGGL(hybrid_gat_kernel, dim3(blocks), dim3(128), 0, stream, p);
}
